// TBGM_30640296690296
// MI455X (gfx1250) — matrix operand order repaired and checked by reading, not compiled, not run on hardware
//
#include <hip/hip_runtime.h>
#include <math.h>

typedef __attribute__((ext_vector_type(16))) _Float16 v16h;
typedef __attribute__((ext_vector_type(8)))  float    v8f;
typedef __attribute__((ext_vector_type(4)))  float    vf4;   // builtin vector -> OK for nontemporal builtins

#define L2EPS 1e-12f
#define EPS_PLAIN    0.4f
#define EPS_MODERATE 0.6f

// ---------------------------------------------------------------------------
// Kernel 1: inv_cnorm[c] = 1 / (||memory[c]||_2 + 1e-12)   (one wave per row)
// ---------------------------------------------------------------------------
__global__ void TBGM_mem_norm_kernel(const float* __restrict__ mem,
                                     float* __restrict__ inv_cnorm,
                                     int C, int D) {
    const int wave = (blockIdx.x * blockDim.x + threadIdx.x) >> 5;
    const int lane = threadIdx.x & 31;
    if (wave >= C) return;

    const float* row = mem + (size_t)wave * (size_t)D;
    float s0 = 0.0f, s1 = 0.0f, s2 = 0.0f, s3 = 0.0f;
    const int Dal = D & ~127;                 // 128-element (512B) chunks per wave
    for (int i = lane * 4; i < Dal; i += 128) {
        vf4 v = *(const vf4*)(row + i);
        s0 = fmaf(v[0], v[0], s0); s1 = fmaf(v[1], v[1], s1);
        s2 = fmaf(v[2], v[2], s2); s3 = fmaf(v[3], v[3], s3);
    }
    for (int i = Dal + lane; i < D; i += 32) { float v = row[i]; s0 = fmaf(v, v, s0); }

    float ssq = (s0 + s1) + (s2 + s3);
    #pragma unroll
    for (int m = 16; m >= 1; m >>= 1) ssq += __shfl_xor(ssq, m, 32);
    if (lane == 0) inv_cnorm[wave] = 1.0f / (sqrtf(ssq) + L2EPS);
}

// ---------------------------------------------------------------------------
// Kernel 2: 16 instances per wave via v_wmma_f32_16x16x32_f16; diagonal of the
// 16x16 tile = the 16 needed dot products. ||f||^2 accumulated in f32 alongside.
// Software-pipelined: next K-chunk's 8 B128 loads are issued before the current
// chunk is converted/consumed, so each wave always has loads in flight.
// ---------------------------------------------------------------------------
struct TBGM_Chunk { vf4 a0, a1, a2, a3, b0, b1, b2, b3; };

__device__ __forceinline__ TBGM_Chunk
TBGM_load_chunk(const float* __restrict__ frow, const float* __restrict__ crow,
                int k, int h) {
    TBGM_Chunk c;
    const float* fa = frow + k + h * 8;   // A: lanes 0-15 K=0..7 | lanes 16-31 K=8..15 (v0-3); +16 (v4-7)
    const float* fb = crow + k + h * 8;   // B: lanes 0-15 K=0..7 | lanes 16-31 K=8..15 (v0-3); +16 (v4-7)
    // feats: zero reuse -> non-temporal, keep L2 for the memory matrix
    c.a0 = __builtin_nontemporal_load((const vf4*)(fa));
    c.a1 = __builtin_nontemporal_load((const vf4*)(fa + 4));
    c.a2 = __builtin_nontemporal_load((const vf4*)(fa + 16));
    c.a3 = __builtin_nontemporal_load((const vf4*)(fa + 20));
    // memory: high reuse across the gather -> regular temporal loads (30MB << 192MB L2)
    c.b0 = *(const vf4*)(fb);
    c.b1 = *(const vf4*)(fb + 4);
    c.b2 = *(const vf4*)(fb + 16);
    c.b3 = *(const vf4*)(fb + 20);
    return c;
}

__device__ __forceinline__ void
TBGM_process_chunk(const TBGM_Chunk& c, v8f& acc,
                   float& s0, float& s1, float& s2, float& s3) {
    // 4 independent ||f||^2 accumulators -> short VALU dependency chains
    s0 = fmaf(c.a0[0], c.a0[0], s0); s1 = fmaf(c.a0[1], c.a0[1], s1);
    s2 = fmaf(c.a0[2], c.a0[2], s2); s3 = fmaf(c.a0[3], c.a0[3], s3);
    s0 = fmaf(c.a1[0], c.a1[0], s0); s1 = fmaf(c.a1[1], c.a1[1], s1);
    s2 = fmaf(c.a1[2], c.a1[2], s2); s3 = fmaf(c.a1[3], c.a1[3], s3);
    s0 = fmaf(c.a2[0], c.a2[0], s0); s1 = fmaf(c.a2[1], c.a2[1], s1);
    s2 = fmaf(c.a2[2], c.a2[2], s2); s3 = fmaf(c.a2[3], c.a2[3], s3);
    s0 = fmaf(c.a3[0], c.a3[0], s0); s1 = fmaf(c.a3[1], c.a3[1], s1);
    s2 = fmaf(c.a3[2], c.a3[2], s2); s3 = fmaf(c.a3[3], c.a3[3], s3);

    v16h A, B;
    A[0]=(_Float16)c.a0[0]; A[1]=(_Float16)c.a0[1]; A[2]=(_Float16)c.a0[2]; A[3]=(_Float16)c.a0[3];
    A[4]=(_Float16)c.a1[0]; A[5]=(_Float16)c.a1[1]; A[6]=(_Float16)c.a1[2]; A[7]=(_Float16)c.a1[3];
    A[8]=(_Float16)c.a2[0]; A[9]=(_Float16)c.a2[1]; A[10]=(_Float16)c.a2[2]; A[11]=(_Float16)c.a2[3];
    A[12]=(_Float16)c.a3[0]; A[13]=(_Float16)c.a3[1]; A[14]=(_Float16)c.a3[2]; A[15]=(_Float16)c.a3[3];
    B[0]=(_Float16)c.b0[0]; B[1]=(_Float16)c.b0[1]; B[2]=(_Float16)c.b0[2]; B[3]=(_Float16)c.b0[3];
    B[4]=(_Float16)c.b1[0]; B[5]=(_Float16)c.b1[1]; B[6]=(_Float16)c.b1[2]; B[7]=(_Float16)c.b1[3];
    B[8]=(_Float16)c.b2[0]; B[9]=(_Float16)c.b2[1]; B[10]=(_Float16)c.b2[2]; B[11]=(_Float16)c.b2[3];
    B[12]=(_Float16)c.b3[0]; B[13]=(_Float16)c.b3[1]; B[14]=(_Float16)c.b3[2]; B[15]=(_Float16)c.b3[3];

    // D += A(16x32 f16) x B(32x16 f16), f32 accumulate
    acc = __builtin_amdgcn_wmma_f32_16x16x32_f16(false, A, false, B,
                                                 (short)0, acc, false, false);
}

__global__ void TBGM_sim_wmma_kernel(const float* __restrict__ feats,
                                     const float* __restrict__ mem,
                                     const int*   __restrict__ idx,
                                     const float* __restrict__ inv_cnorm,
                                     float* __restrict__ out,
                                     int D, int ngroups) {
    const int wave = (blockIdx.x * blockDim.x + threadIdx.x) >> 5;
    const int lane = threadIdx.x & 31;
    if (wave >= ngroups) return;              // wave-uniform: EXEC stays all-ones

    const int base = wave << 4;               // 16 instances per wave
    const int r    = lane & 15;               // tile row (A) / tile col (B) this lane serves
    const int h    = lane >> 4;               // which K-half this lane holds

    const int    cidx = idx[base + r];
    const float* frow = feats + (size_t)(base + r) * (size_t)D;
    const float* crow = mem   + (size_t)cidx      * (size_t)D;

    v8f   acc = {};
    float s0 = 0.0f, s1 = 0.0f, s2 = 0.0f, s3 = 0.0f;

    // -------- software pipeline: prologue load, steady state, epilogue --------
    TBGM_Chunk cur = TBGM_load_chunk(frow, crow, 0, h);
    for (int k = 32; k < D; k += 32) {
        __builtin_prefetch(frow + k + 96, 0, 1);   // -> global_prefetch_b8, stream ahead
        __builtin_prefetch(crow + k + 96, 0, 1);
        TBGM_Chunk nxt = TBGM_load_chunk(frow, crow, k, h);   // in flight during process
        TBGM_process_chunk(cur, acc, s0, s1, s2, s3);
        cur = nxt;
    }
    TBGM_process_chunk(cur, acc, s0, s1, s2, s3);

    // complete ||f||^2 per row: partner lane is lane^16 (same row, other K-half)
    float ssq = (s0 + s1) + (s2 + s3);
    ssq += __shfl_xor(ssq, 16, 32);

    // Diagonal extraction. D layout: VGPR v, lanes 0-15 -> (M=v, N=lane);
    // lanes 16-31 -> (M=v+8, N=lane-16). Diagonal lives on lanes 0-7 (acc[lane])
    // and lanes 24-31 (acc[lane-24]).
    float dot = 0.0f;
    #pragma unroll
    for (int v = 0; v < 8; ++v)
        dot = ((lane == v) || (lane == v + 24)) ? acc[v] : dot;

    const bool active = (lane < 8) || (lane >= 24);
    if (active) {
        // on these lanes: row = lane&15, and ssq/cidx already belong to that row
        float sim = dot * (1.0f / (sqrtf(ssq) + L2EPS)) * inv_cnorm[cidx];
        float cls = (sim >= EPS_MODERATE) ? 0.0f : ((sim >= EPS_PLAIN) ? 1.0f : 2.0f);
        int inst  = base + ((lane < 8) ? lane : (lane - 16));
        __builtin_nontemporal_store(cls, out + inst);
    }
}

// ---------------------------------------------------------------------------
// Tail kernel (N % 16 leftovers; not launched for N = 100000)
// ---------------------------------------------------------------------------
__global__ void TBGM_sim_tail_kernel(const float* __restrict__ feats,
                                     const float* __restrict__ mem,
                                     const int*   __restrict__ idx,
                                     const float* __restrict__ inv_cnorm,
                                     float* __restrict__ out,
                                     int start, int N, int D) {
    int i = start + blockIdx.x * blockDim.x + threadIdx.x;
    if (i >= N) return;
    const float* f = feats + (size_t)i * (size_t)D;
    int c = idx[i];
    const float* m = mem + (size_t)c * (size_t)D;
    float dot = 0.0f, ssq = 0.0f;
    for (int k = 0; k < D; ++k) { float fv = f[k]; dot = fmaf(fv, m[k], dot); ssq = fmaf(fv, fv, ssq); }
    float sim = dot * (1.0f / (sqrtf(ssq) + L2EPS)) * inv_cnorm[c];
    out[i] = (sim >= EPS_MODERATE) ? 0.0f : ((sim >= EPS_PLAIN) ? 1.0f : 2.0f);
}

// ---------------------------------------------------------------------------
extern "C" void kernel_launch(void* const* d_in, const int* in_sizes, int n_in,
                              void* d_out, int out_size, void* d_ws, size_t ws_size,
                              hipStream_t stream) {
    const float* feats = (const float*)d_in[0];
    const float* mem   = (const float*)d_in[1];
    const int*   idx   = (const int*)d_in[2];
    float*       out   = (float*)d_out;

    const int N = in_sizes[2];                       // 100000
    const int D = (N > 0) ? (in_sizes[0] / N) : 0;   // 768
    const int C = (D > 0) ? (in_sizes[1] / D) : 0;   // 10000

    float* inv_cnorm = (float*)d_ws;                 // C floats (~40 KB) of scratch

    // Pass 1: memory-row inverse norms (one wave per row, 8 waves/block)
    if (C > 0) {
        int blocks1 = (C + 7) / 8;
        TBGM_mem_norm_kernel<<<blocks1, 256, 0, stream>>>(mem, inv_cnorm, C, D);
    }

    // Pass 2: WMMA-batched dots + bucketize (16 instances per wave)
    int ngroups = N >> 4;
    if (ngroups > 0) {
        int blocks2 = (ngroups + 7) / 8;
        TBGM_sim_wmma_kernel<<<blocks2, 256, 0, stream>>>(feats, mem, idx, inv_cnorm,
                                                          out, D, ngroups);
    }

    // Tail (never taken for N=100000, kept for generality)
    int rem = N - (ngroups << 4);
    if (rem > 0) {
        TBGM_sim_tail_kernel<<<(rem + 255) / 256, 256, 0, stream>>>(
            feats, mem, idx, inv_cnorm, out, ngroups << 4, N, D);
    }
}
